// RevolutionNaive_91319594647915
// MI455X (gfx1250) — compile-verified, matrix operand order checked by reading, not run on hardware
//
#include <hip/hip_runtime.h>
#include <hip/hip_bf16.h>

// ---------------------------------------------------------------------------
// Fused "Revolution" operator for MI455X (gfx1250), wave32 + WMMA bf16.
//
// Shapes: B=8, C=256, G=8, GC=32, K=3, H=W=56, NEW=2
//   feat:  1608 per position (pad to 1632 = 51 * 32 K-steps)
//   GEMM1: [16 pos x 1632] @ [1632 x 80]  (72 valid outs) -> 255 wmma / WG
//   GEMM2: [16 pos x 96]   @ [96 x 288]                   ->  54 wmma / WG
// Phase-0 halo staging uses GLOBAL_LOAD_ASYNC_TO_LDS_B32 (ASYNCcnt path).
// ---------------------------------------------------------------------------

typedef __attribute__((ext_vector_type(16))) __bf16 v16bf;
typedef __attribute__((ext_vector_type(8)))  float  v8f;

#define CC   256
#define HH   56
#define WW   56
#define IN1  1608
#define KT1  51        // ceil(1608/32)
#define NT1  5         // ceil(72/16)
#define OUT1 72
#define KT2  3         // ceil(72/32) padded to 96
#define NT2  18        // 288/16
#define OUT2 288
#define FRAG_ELEMS 512 // 32 lanes * 16 bf16 per 16x32 (or 32x16) fragment

#define LDS_IN_FLOATS (CC * 3 * 20)                 // 15360 f32 = 61440 B (col stride 20)
#define LDS_IN_BYTES  (LDS_IN_FLOATS * 4)
#define FEAT_BYTES    (KT1 * FRAG_ELEMS * 2)        // 52224 B (union with y2: 16*288*4=18432 B)
#define Y1F_BYTES     (KT2 * FRAG_ELEMS * 2)        // 3072 B
#define SMEM_BYTES    (LDS_IN_BYTES + FEAT_BYTES + Y1F_BYTES)  // 116736 B

// element i (0..15) of a 16-bit WMMA fragment maps to K = (i<8 ? i : i+8) + 8*(lane>=16)
__device__ __forceinline__ int frag_k(int lane, int i) {
    return ((i < 8) ? i : i + 8) + ((lane >> 4) << 3);
}

// ---------------------------------------------------------------------------
// Pre-swizzle W1 (72 x 1608 f32) into bf16 B-fragments: [kt][nt][lane][elem]
// ---------------------------------------------------------------------------
__global__ void pack_w1_kernel(const float* __restrict__ W1, __bf16* __restrict__ W1p) {
    int idx = blockIdx.x * 256 + threadIdx.x;
    if (idx >= KT1 * NT1 * FRAG_ELEMS) return;
    int kt = idx / (NT1 * FRAG_ELEMS);
    int r  = idx % (NT1 * FRAG_ELEMS);
    int nt = r / FRAG_ELEMS;
    int q  = r % FRAG_ELEMS;
    int l  = q >> 4, i = q & 15;
    int n  = nt * 16 + (l & 15);
    int k  = kt * 32 + frag_k(l, i);
    float v = (n < OUT1 && k < IN1) ? W1[n * IN1 + k] : 0.0f;
    W1p[idx] = (__bf16)v;
}

// ---------------------------------------------------------------------------
// Pre-swizzle W2 (288 x 72 f32) into bf16 B-fragments: [kt][nt][lane][elem]
// ---------------------------------------------------------------------------
__global__ void pack_w2_kernel(const float* __restrict__ W2, __bf16* __restrict__ W2p) {
    int idx = blockIdx.x * 256 + threadIdx.x;
    if (idx >= KT2 * NT2 * FRAG_ELEMS) return;
    int kt = idx / (NT2 * FRAG_ELEMS);
    int r  = idx % (NT2 * FRAG_ELEMS);
    int nt = r / FRAG_ELEMS;
    int q  = r % FRAG_ELEMS;
    int l  = q >> 4, i = q & 15;
    int n  = nt * 16 + (l & 15);
    int k  = kt * 32 + frag_k(l, i);
    float v = (k < OUT1) ? W2[n * OUT1 + k] : 0.0f;
    W2p[idx] = (__bf16)v;
}

// ---------------------------------------------------------------------------
// Main fused kernel. Grid: (4 w-tiles, 56 h, 8 b). Block: 256 threads (8 wave32).
// ---------------------------------------------------------------------------
__global__ void revolution_fused_kernel(const float* __restrict__ inp,
                                        const __bf16* __restrict__ W1p,
                                        const float* __restrict__ b1,
                                        const __bf16* __restrict__ W2p,
                                        const float* __restrict__ b2,
                                        float* __restrict__ out) {
    extern __shared__ __attribute__((aligned(32))) char smem_raw[];
    float*  lds_in = (float*)smem_raw;                               // [256][3][20]
    __bf16* featF  = (__bf16*)(smem_raw + LDS_IN_BYTES);             // [51][512] A-frags
    float*  y2     = (float*)(smem_raw + LDS_IN_BYTES);              // [16][288] (union w/ featF)
    __bf16* y1F    = (__bf16*)(smem_raw + LDS_IN_BYTES + FEAT_BYTES);// [3][512]  A-frags

    const int bb = blockIdx.z, h = blockIdx.y, w0 = blockIdx.x * 16;
    const int tid  = threadIdx.x;
    const int wave = tid >> 5, lane = tid & 31;

    // --- Phase 0: zero y1F pad region + async-stage 3x18 halo for all 256 channels ---
    for (int i = tid; i < KT2 * FRAG_ELEMS; i += 256) y1F[i] = (__bf16)0.0f;

    // Dynamic LDS begins at LDS offset 0 (no static __shared__ in this kernel),
    // so layout byte offsets are valid LDS addresses for the async-DMA path.
    for (int e = tid; e < CC * 3 * 18; e += 256) {
        int c = e / 54, r = e % 54;
        int ki = r / 18, cj = r % 18;
        int row = h + ki - 1, col = w0 + cj - 1;
        unsigned ldsb = (unsigned)(((c * 3 + ki) * 20 + cj) * 4);   // LDS byte offset
        if (row >= 0 && row < HH && col >= 0 && col < WW) {
            const float* gp = &inp[(((size_t)bb * CC + c) * HH + row) * WW + col];
            unsigned long long ga = (unsigned long long)gp;
            // GLOBAL_LOAD_ASYNC_TO_LDS_B32: copies one dword from global memory to LDS byte offset ldsb; tracked by ASYNCcnt
            asm volatile("global_load_async_to_lds_b32 %0, %1, off"
                         :: "v"(ldsb), "v"(ga) : "memory");
        } else {
            lds_in[(c * 3 + ki) * 20 + cj] = 0.0f;                  // zero-pad halo
        }
    }
    asm volatile("s_wait_asynccnt 0x0" ::: "memory");
    __syncthreads();

    // --- Phase 1: build feat directly in WMMA A-fragment order (bf16) ---
    // fragment elem (kt, lane l, elem i): row m = l&15 (= position p), K = kt*32 + frag_k(l,i)
    for (int e = tid; e < KT1 * FRAG_ELEMS; e += 256) {
        int kt = e >> 9, q = e & 511;
        int l = q >> 4, i = q & 15;
        int p = l & 15;
        int k = kt * 32 + frag_k(l, i);
        float v = 0.0f;
        if (k < 72) {                       // x1: per-group max over 32 channels
            int g = k / 9, r = k % 9, ki = r / 3, kj = r % 3;
            const float* base = &lds_in[((g * 32) * 3 + ki) * 20 + p + kj];
            float m = -3.4e38f;
            #pragma unroll
            for (int c = 0; c < 32; ++c) m = fmaxf(m, base[c * 60]);
            v = m;
        } else if (k < 840) {               // x2: max over kernel rows (ki)
            int t = k - 72, ch = t / 3, kj = t % 3;
            const float* base = &lds_in[(ch * 3) * 20 + p + kj];
            v = fmaxf(fmaxf(base[0], base[20]), base[40]);
        } else if (k < IN1) {               // x3: max over kernel cols (kj)
            int t = k - 840, ch = t / 3, ki = t % 3;
            const float* base = &lds_in[(ch * 3 + ki) * 20 + p];
            v = fmaxf(fmaxf(base[0], base[1]), base[2]);
        }
        featF[e] = (__bf16)v;
    }
    __syncthreads();

    // --- Phase 2: GEMM1 [16x1632] @ [1632x80] via v_wmma_f32_16x16x32_bf16 ---
    if (wave < NT1) {
        const int nt = wave;
        v8f acc = {};
        for (int kt = 0; kt < KT1; ++kt) {
            v16bf a  = *(const v16bf*)(featF + kt * FRAG_ELEMS + lane * 16);
            const __bf16* bp = W1p + (size_t)(kt * NT1 + nt) * FRAG_ELEMS + lane * 16;
            if (kt + 1 < KT1)   // WGP-scope prefetch of next B fragment
                __builtin_prefetch(W1p + (size_t)((kt + 1) * NT1 + nt) * FRAG_ELEMS + lane * 16, 0, 3);
            v16bf bv = *(const v16bf*)bp;
            acc = __builtin_amdgcn_wmma_f32_16x16x32_bf16(false, a, false, bv,
                                                          (short)0, acc, false, false);
        }
        // bias + re-swizzle y1 (pos x 72, zero-pad to 96) into A-fragments for GEMM2
        const int n = lane & 15, o = nt * 16 + n, hiL = lane >> 4;
        const float bias = (o < OUT1) ? b1[o] : 0.0f;
        #pragma unroll
        for (int v = 0; v < 8; ++v) {
            int m = v + 8 * hiL;            // position p
            float val = (o < OUT1) ? (acc[v] + bias) : 0.0f;
            int kin = o & 31, kt2 = o >> 5;
            int hib = (kin >> 3) & 1;
            int i2 = ((kin >> 4) << 3) | (kin & 7);
            int lane2 = m + (hib << 4);
            y1F[kt2 * FRAG_ELEMS + lane2 * 16 + i2] = (__bf16)val;
        }
    }
    __syncthreads();

    // --- Phase 3: GEMM2 [16x96] @ [96x288] -> y2 in LDS (f32, overwrites featF) ---
    for (int nt2 = wave; nt2 < NT2; nt2 += 8) {
        v8f acc = {};
        #pragma unroll
        for (int kt2 = 0; kt2 < KT2; ++kt2) {
            v16bf a  = *(const v16bf*)(y1F + kt2 * FRAG_ELEMS + lane * 16);
            v16bf bv = *(const v16bf*)(W2p + (size_t)(kt2 * NT2 + nt2) * FRAG_ELEMS + lane * 16);
            acc = __builtin_amdgcn_wmma_f32_16x16x32_bf16(false, a, false, bv,
                                                          (short)0, acc, false, false);
        }
        const int n = lane & 15, o2 = nt2 * 16 + n, hiL = lane >> 4;
        const float bias = b2[o2];
        #pragma unroll
        for (int v = 0; v < 8; ++v) {
            int p = v + 8 * hiL;
            y2[p * OUT2 + o2] = acc[v] + bias;
        }
    }
    __syncthreads();

    // --- Phase 4a: softmax over the 9 kernel taps, in place in y2 ---
    for (int t = tid; t < 512; t += 256) {      // (p, g, n): 16*8*4
        int p = t >> 5, r = t & 31, g = r >> 2, n = r & 3;
        float* yb = &y2[p * OUT2 + g * 36 + n];
        float vals[9];
        float mx = -3.4e38f;
        #pragma unroll
        for (int kk = 0; kk < 9; ++kk) { vals[kk] = yb[kk * 4]; mx = fmaxf(mx, vals[kk]); }
        float s = 0.0f;
        #pragma unroll
        for (int kk = 0; kk < 9; ++kk) { vals[kk] = __expf(vals[kk] - mx); s += vals[kk]; }
        float inv = __frcp_rn(s);
        #pragma unroll
        for (int kk = 0; kk < 9; ++kk) yb[kk * 4] = vals[kk] * inv;
    }
    __syncthreads();

    // --- Phase 4b: softmax-weighted 3x3 aggregation + coalesced float2 stores ---
    const int valid = (WW - w0 < 16) ? (WW - w0) : 16;
    const int p = tid & 15, chb = tid >> 4;
    for (int pass = 0; pass < 16; ++pass) {
        int ch = pass * 16 + chb;
        int g = ch >> 5;
        float patch[9];
        #pragma unroll
        for (int ki = 0; ki < 3; ++ki)
            #pragma unroll
            for (int kj = 0; kj < 3; ++kj)
                patch[ki * 3 + kj] = lds_in[(ch * 3 + ki) * 20 + p + kj];
        const float* wb = &y2[p * OUT2 + g * 36];
        float s0 = 0.f, s1 = 0.f, s2 = 0.f, s3 = 0.f;
        #pragma unroll
        for (int kk = 0; kk < 9; ++kk) {
            float pv = patch[kk];
            s0 = fmaf(pv, wb[kk * 4 + 0], s0);
            s1 = fmaf(pv, wb[kk * 4 + 1], s1);
            s2 = fmaf(pv, wb[kk * 4 + 2], s2);
            s3 = fmaf(pv, wb[kk * 4 + 3], s3);
        }
        if (p < valid) {
            int wc = 2 * (w0 + p);
            size_t base = (((size_t)bb * CC + ch) * 112 + 2 * h) * 112 + wc;
            *(float2*)&out[base]       = make_float2(s0, s1);
            *(float2*)&out[base + 112] = make_float2(s2, s3);
        }
    }
}

// ---------------------------------------------------------------------------
extern "C" void kernel_launch(void* const* d_in, const int* in_sizes, int n_in,
                              void* d_out, int out_size, void* d_ws, size_t ws_size,
                              hipStream_t stream) {
    const float* inp = (const float*)d_in[0];
    const float* W1  = (const float*)d_in[1];
    const float* b1  = (const float*)d_in[2];
    const float* W2  = (const float*)d_in[3];
    const float* b2  = (const float*)d_in[4];
    float* out = (float*)d_out;

    __bf16* W1p = (__bf16*)d_ws;                               // 51*5*512 bf16
    __bf16* W2p = W1p + (size_t)KT1 * NT1 * FRAG_ELEMS;        // 3*18*512 bf16

    // Idempotent, host-side, capture-safe: no static guard (harness requirement).
    hipFuncSetAttribute((const void*)revolution_fused_kernel,
                        hipFuncAttributeMaxDynamicSharedMemorySize, SMEM_BYTES);

    int n1 = KT1 * NT1 * FRAG_ELEMS;
    int n2 = KT2 * NT2 * FRAG_ELEMS;
    pack_w1_kernel<<<(n1 + 255) / 256, 256, 0, stream>>>(W1, W1p);
    pack_w2_kernel<<<(n2 + 255) / 256, 256, 0, stream>>>(W2, W2p);

    dim3 grid(4, HH, 8);            // 4 w-tiles of 16 (last has 8 valid), 56 rows, 8 batches
    revolution_fused_kernel<<<grid, 256, SMEM_BYTES, stream>>>(inp, W1p, b1, W2p, b2, out);
}
